// GraphAttentionLayer_31473520345705
// MI455X (gfx1250) — matrix operand order repaired and checked by reading, not compiled, not run on hardware
//
#include <hip/hip_runtime.h>
#include <math.h>

#define N_HID 128
#define LSTR  132                 // padded LDS row stride (floats) to avoid bank conflicts
#define SCARE 11.313708498984761f // sqrt(128)
#define NEGV  (-9.0e15f)

typedef __attribute__((ext_vector_type(16))) __bf16 v16bf;
typedef __attribute__((ext_vector_type(8)))  float  v8f;

// A-operand fragment (16-bit A 16x32 layout): lane holds row, elements e:0..7 -> k0+e,
// e:8..15 -> k0+16+(e-8).  p points at (row, k0).
__device__ __forceinline__ v16bf fragA_runs8(const float* p) {
  v16bf f;
#pragma unroll
  for (int e = 0; e < 8; ++e) f[e] = (__bf16)p[e];
#pragma unroll
  for (int e = 0; e < 8; ++e) f[8 + e] = (__bf16)p[16 + e];
  return f;
}

// Same A fragment but from global memory with vectorized 128-bit loads.
__device__ __forceinline__ v16bf fragA_runs8_g(const float* p) {
  const float4* q = (const float4*)p;   // 32B-aligned (row 512B aligned, k0 multiple of 8)
  float4 a = q[0], b = q[1], c = q[4], d = q[5];
  v16bf f;
  f[0]=(__bf16)a.x; f[1]=(__bf16)a.y; f[2] =(__bf16)a.z; f[3] =(__bf16)a.w;
  f[4]=(__bf16)b.x; f[5]=(__bf16)b.y; f[6] =(__bf16)b.z; f[7] =(__bf16)b.w;
  f[8]=(__bf16)c.x; f[9]=(__bf16)c.y; f[10]=(__bf16)c.z; f[11]=(__bf16)c.w;
  f[12]=(__bf16)d.x;f[13]=(__bf16)d.y;f[14]=(__bf16)d.z; f[15]=(__bf16)d.w;
  return f;
}

// B-operand fragment (16-bit B 32x16 layout): lane holds column, elements e:0..7 -> k0+e,
// e:8..15 -> k0+16+(e-8).  p points at (col, k0).
__device__ __forceinline__ v16bf fragB_run16(const float* p) {
  v16bf f;
#pragma unroll
  for (int e = 0; e < 8; ++e) f[e] = (__bf16)p[e];
#pragma unroll
  for (int e = 0; e < 8; ++e) f[8 + e] = (__bf16)p[16 + e];
  return f;
}

__device__ __forceinline__ v8f wmma_bf16(v16bf a, v16bf b, v8f c) {
  return __builtin_amdgcn_wmma_f32_16x16x32_bf16(false, a, false, b, (short)0, c, false, false);
}

// ---------------------------------------------------------------------------
// Pack Q,K,V (f32 128x128 row-major) into bf16 WMMA B-operand fragments.
// ws layout (bf16 units): [w][t*4+s][lane][e]  -> gid = w*16384 + frag*512 + lane*16 + e
// mapping: h = 16*t + lane%16 ; k = 32*s + 8*(lane/16) + (e<8 ? e : 8+e)
// ---------------------------------------------------------------------------
__global__ void pack_weights_kernel(const float* __restrict__ Q, const float* __restrict__ K,
                                    const float* __restrict__ V, unsigned short* __restrict__ ws) {
  int gid = blockIdx.x * blockDim.x + threadIdx.x;
  if (gid >= 3 * 16384) return;
  int w    = gid >> 14;
  int idx  = gid & 16383;
  int frag = idx >> 9;         // t*4+s, 0..31
  int lane = (idx >> 4) & 31;
  int e    = idx & 15;
  int t = frag >> 2, s = frag & 3;
  int h = 16 * t + (lane & 15);
  int k = 32 * s + ((lane >> 4) << 3) + (e < 8 ? e : 8 + e);
  const float* W = (w == 0) ? Q : (w == 1) ? K : V;
  union { __bf16 b; unsigned short s; } cv; cv.b = (__bf16)W[k * N_HID + h];
  ws[gid] = cv.s;
}

// ---------------------------------------------------------------------------
// One wave32 per node.
// ---------------------------------------------------------------------------
__global__ __launch_bounds__(32) void gat_node_kernel(
    const float* __restrict__ x, const int* __restrict__ nbl,
    const unsigned short* __restrict__ wfrag_raw,
    float* __restrict__ out, int n)
{
  __shared__ float M0[16 * LSTR];  // Qn  -> later Kc' (=nres@K)
  __shared__ float M1[16 * LSTR];  // Kn  -> later Vc' (=nres@V)
  __shared__ float M2[16 * LSTR];  // Vn
  __shared__ float M3[16 * LSTR];  // nres
  __shared__ float Ssc[256];       // s then a (16x16)
  __shared__ float Xi[N_HID], Qx[N_HID], Kx[N_HID], Vx[N_HID];
  __shared__ float S2[32];         // s2 then a2 (17)
  __shared__ float Msk[16];
  __shared__ int   Nbr[16];
  __shared__ float Red[32];

  const v16bf* wfrag = (const v16bf*)wfrag_raw;
  const int node = blockIdx.x;
  if (node >= n) return;
  const int lane = threadIdx.x;
  const int r = lane & 15, half = lane >> 4;

  if (lane < 16) {
    int nb = nbl[node * 16 + lane];
    Nbr[lane] = nb;
    Msk[lane] = (nb != (n - 1)) ? 1.0f : 0.0f;
  }
#pragma unroll
  for (int c = 0; c < 4; ++c)
    Xi[lane + 32 * c] = x[(size_t)node * N_HID + lane + 32 * c];
  __syncthreads();

  // ---- A fragments: gathered neighbor rows, and broadcast x_i (all 16 rows = x_i) ----
  const float* xrow = x + (size_t)Nbr[r] * N_HID;
  v16bf afrag[4], xfrag[4];
#pragma unroll
  for (int s = 0; s < 4; ++s) {
    afrag[s] = fragA_runs8_g(xrow + 32 * s + 8 * half);
    xfrag[s] = fragA_runs8(&Xi[32 * s + 8 * half]);
  }

  // ---- Stage 1: {Qn,Kn,Vn} = nm @ {Q,K,V} and {Qx,Kx,Vx} = x_i @ {Q,K,V}
  //      (192 WMMAs, each loaded B fragment feeds two WMMAs) ----
  float* const mats[3]  = {M0, M1, M2};
  float* const xvecs[3] = {Qx, Kx, Vx};
#pragma unroll
  for (int w = 0; w < 3; ++w) {
#pragma unroll
    for (int t = 0; t < 8; ++t) {
      v16bf bfr[4];
#pragma unroll
      for (int s = 0; s < 4; ++s)
        bfr[s] = wfrag[w * 1024 + (t * 4 + s) * 32 + lane];
      v8f accN = {}, accX = {};
#pragma unroll
      for (int s = 0; s < 4; ++s) {
        accN = wmma_bf16(afrag[s], bfr[s], accN);
        accX = wmma_bf16(xfrag[s], bfr[s], accX);
      }
#pragma unroll
      for (int d = 0; d < 8; ++d)
        mats[w][(d + 8 * half) * LSTR + 16 * t + r] = accN[d];  // C layout: M=d+8*half, N=r
      if (half == 0)
        xvecs[w][16 * t + r] = accX[0];  // all rows identical (broadcast A)
    }
  }
  __syncthreads();

  // ---- s = Qn @ Kn^T (4 WMMAs; B[k][n] = Kn[n][k] -> contiguous Kn row reads) ----
  {
    v8f acc = {};
#pragma unroll
    for (int s = 0; s < 4; ++s) {
      v16bf aq = fragA_runs8(&M0[r * LSTR + 32 * s + 8 * half]);
      v16bf bk = fragB_run16(&M1[r * LSTR + 32 * s + 8 * half]);
      acc = wmma_bf16(aq, bk, acc);
    }
#pragma unroll
    for (int d = 0; d < 8; ++d) {
      int i = d + 8 * half;
      float v = acc[d];
      v = (v > 0.f) ? v : 0.01f * v;                    // leaky_relu
      v = (Msk[i] * Msk[r] > 0.f) ? v : NEGV;           // mask1
      Ssc[i * 16 + r] = v * SCARE;                      // * sqrt(nhid)
    }
  }
  __syncthreads();

  // ---- row softmax over 16 (lanes 0..15, one row each) ----
  if (lane < 16) {
    float mx = -3.4e38f;
#pragma unroll
    for (int j = 0; j < 16; ++j) mx = fmaxf(mx, Ssc[lane * 16 + j]);
    float sum = 0.f;
#pragma unroll
    for (int j = 0; j < 16; ++j) sum += __expf(Ssc[lane * 16 + j] - mx);
    float inv = 1.f / sum;
#pragma unroll
    for (int j = 0; j < 16; ++j) Ssc[lane * 16 + j] = __expf(Ssc[lane * 16 + j] - mx) * inv;
  }
  __syncthreads();

  // ---- nres = a @ Vn (8 WMMAs, K padded 16->32 with zeros) ----
  {
    v16bf fa;
#pragma unroll
    for (int e = 0; e < 16; ++e) fa[e] = (__bf16)(half == 0 ? Ssc[r * 16 + e] : 0.f);  // k = 16*half+e; k >= 16 padded
#pragma unroll
    for (int t = 0; t < 8; ++t) {
      v16bf fb;
      if (half == 0) {
#pragma unroll
        for (int e = 0; e < 16; ++e) fb[e] = (__bf16)M2[e * LSTR + 16 * t + r];  // k = e
      } else {
#pragma unroll
        for (int e = 0; e < 16; ++e) fb[e] = (__bf16)0.f;                        // k >= 16 padded
      }
      v8f acc = {};
      acc = wmma_bf16(fa, fb, acc);
#pragma unroll
      for (int d = 0; d < 8; ++d)
        M3[(d + 8 * half) * LSTR + 16 * t + r] = acc[d];
    }
  }
  __syncthreads();

  // ---- Stage 2: Kc' = nres@K -> M0, Vc' = nres@V -> M1 (64 WMMAs) ----
  v16bf a2f[4];
#pragma unroll
  for (int s = 0; s < 4; ++s)
    a2f[s] = fragA_runs8(&M3[r * LSTR + 32 * s + 8 * half]);
#pragma unroll
  for (int w = 1; w < 3; ++w) {
    float* dst = (w == 1) ? M0 : M1;
#pragma unroll
    for (int t = 0; t < 8; ++t) {
      v16bf bfr[4];
#pragma unroll
      for (int s = 0; s < 4; ++s)
        bfr[s] = wfrag[w * 1024 + (t * 4 + s) * 32 + lane];
      v8f acc = {};
#pragma unroll
      for (int s = 0; s < 4; ++s)
        acc = wmma_bf16(a2f[s], bfr[s], acc);
#pragma unroll
      for (int d = 0; d < 8; ++d)
        dst[(d + 8 * half) * LSTR + 16 * t + r] = acc[d];
    }
  }
  __syncthreads();

  // ---- s2 rows 1..16 = Qx . Kc'_j via WMMA (A = broadcast Qx, B = Kc'^T) ----
  {
    v8f acc = {};
#pragma unroll
    for (int s = 0; s < 4; ++s) {
      v16bf aq = fragA_runs8(&Qx[32 * s + 8 * half]);
      v16bf bk = fragB_run16(&M0[r * LSTR + 32 * s + 8 * half]);
      acc = wmma_bf16(aq, bk, acc);
    }
    // s2[0] = Qx.Kx partials (4 cols/lane)
    float p0 = 0.f;
#pragma unroll
    for (int c = 0; c < 4; ++c) {
      int h = lane + 32 * c;
      p0 = fmaf(Qx[h], Kx[h], p0);
    }
    Red[lane] = p0;
    if (half == 0) {
      float v = acc[0] * SCARE;                      // scale first (reference order)
      v = (v > 0.f) ? v : 0.01f * v;                 // then leaky
      S2[1 + r] = (Msk[r] > 0.f) ? v : NEGV;         // then mask2
    }
  }
  __syncthreads();
  if (lane == 0) {
    float v0 = 0.f;
    for (int i = 0; i < 32; ++i) v0 += Red[i];
    v0 *= SCARE;
    v0 = (v0 > 0.f) ? v0 : 0.01f * v0;
    S2[0] = v0;                                      // mask2[0] == 1 always
    float mx = -3.4e38f;
    for (int j = 0; j < 17; ++j) mx = fmaxf(mx, S2[j]);
    float sum = 0.f;
    for (int j = 0; j < 17; ++j) sum += __expf(S2[j] - mx);
    float inv = 1.f / sum;
    for (int j = 0; j < 17; ++j) S2[j] = __expf(S2[j] - mx) * inv;
  }
  __syncthreads();

  // ---- out = a2 @ [Vx; Vc'], then L2 normalize ----
  float o[4];
  float nrm = 0.f;
#pragma unroll
  for (int c = 0; c < 4; ++c) {
    int h = lane + 32 * c;
    float acc = S2[0] * Vx[h];
#pragma unroll
    for (int j = 1; j < 17; ++j) acc = fmaf(S2[j], M1[(j - 1) * LSTR + h], acc);
    o[c] = acc;
    nrm = fmaf(acc, acc, nrm);
  }
  Red[lane] = nrm;
  __syncthreads();
  if (lane == 0) {
    float t = 0.f;
    for (int i = 0; i < 32; ++i) t += Red[i];
    Red[0] = 1.f / fmaxf(sqrtf(t), 1e-12f);
  }
  __syncthreads();
  float inv = Red[0];
#pragma unroll
  for (int c = 0; c < 4; ++c)
    out[(size_t)node * N_HID + lane + 32 * c] = o[c] * inv;
}

extern "C" void kernel_launch(void* const* d_in, const int* in_sizes, int n_in,
                              void* d_out, int out_size, void* d_ws, size_t ws_size,
                              hipStream_t stream) {
  (void)n_in; (void)out_size; (void)ws_size;
  const float* x   = (const float*)d_in[0];
  const int*   nbl = (const int*)d_in[1];
  const float* Q   = (const float*)d_in[2];
  const float* K   = (const float*)d_in[3];
  const float* V   = (const float*)d_in[4];
  int n = in_sizes[0] / N_HID;   // 50000

  // Repack Q,K,V into bf16 WMMA B-fragments in workspace (96 KB), every call.
  pack_weights_kernel<<<(3 * 16384 + 255) / 256, 256, 0, stream>>>(
      Q, K, V, (unsigned short*)d_ws);

  gat_node_kernel<<<n, 32, 0, stream>>>(
      x, nbl, (const unsigned short*)d_ws, (float*)d_out, n);
}
